// ConditionedAttnGRUDecoder_52347061403748
// MI455X (gfx1250) — matrix operand order repaired and checked by reading, not compiled, not run on hardware
//
#include <hip/hip_runtime.h>

#define HD   768
#define H3   2304
#define VOC  32000
#define TFD  1024
#define BB   4
#define TT   512
#define ROWS 2048   /* TT*BB */

typedef __bf16 bf16;
typedef bf16  v16bf __attribute__((ext_vector_type(16)));
typedef float v8f   __attribute__((ext_vector_type(8)));
typedef unsigned int v4u __attribute__((ext_vector_type(4)));

union FragU { v16bf v; v4u u[2]; };

__device__ __forceinline__ unsigned short f2bf(float x) {
  unsigned u = __float_as_uint(x);
  u += 0x7FFFu + ((u >> 16) & 1u);       // round-to-nearest-even
  return (unsigned short)(u >> 16);
}

// A fragment at precomputed lane base: pa = A + (m0 + (lane&15))*lda + 16*(lane>>4)
// 16 contiguous K starting at k0 relative to the lane base.
__device__ __forceinline__ v16bf frag_a_at(const unsigned short* pa, int k0) {
  FragU f;
  f.u[0] = *(const v4u*)(pa + k0);
  f.u[1] = *(const v4u*)(pa + k0 + 8);
  return f.v;
}

// B fragment at precomputed lane base: pb = Bm + (n0 + (lane&15))*ldb + 16*(lane>>4)
// 16 contiguous K starting at k0 relative to the lane base.
__device__ __forceinline__ v16bf frag_b_at(const unsigned short* pb, int k0) {
  FragU f;
  f.u[0] = *(const v4u*)(pb + k0);
  f.u[1] = *(const v4u*)(pb + k0 + 8);
  return f.v;
}

// Software-pipelined K-loop over HD, 4 N-tiles sharing one A fragment.
__device__ __forceinline__ void wmma_rowtile4(const unsigned short* pa,
                                              const unsigned short* pb0, const unsigned short* pb1,
                                              const unsigned short* pb2, const unsigned short* pb3,
                                              v8f acc[4]) {
  v16bf a_c  = frag_a_at(pa, 0);
  v16bf b_c0 = frag_b_at(pb0, 0), b_c1 = frag_b_at(pb1, 0);
  v16bf b_c2 = frag_b_at(pb2, 0), b_c3 = frag_b_at(pb3, 0);
#pragma unroll
  for (int k0 = 0; k0 < HD; k0 += 32) {
    v16bf a_n = {}, b_n0 = {}, b_n1 = {}, b_n2 = {}, b_n3 = {};
    if (k0 + 32 < HD) {                       // prefetch next K-slice into fresh regs
      a_n  = frag_a_at(pa, k0 + 32);
      b_n0 = frag_b_at(pb0, k0 + 32); b_n1 = frag_b_at(pb1, k0 + 32);
      b_n2 = frag_b_at(pb2, k0 + 32); b_n3 = frag_b_at(pb3, k0 + 32);
    }
    acc[0] = __builtin_amdgcn_wmma_f32_16x16x32_bf16(false, a_c, false, b_c0, (short)0, acc[0], false, false);
    acc[1] = __builtin_amdgcn_wmma_f32_16x16x32_bf16(false, a_c, false, b_c1, (short)0, acc[1], false, false);
    acc[2] = __builtin_amdgcn_wmma_f32_16x16x32_bf16(false, a_c, false, b_c2, (short)0, acc[2], false, false);
    acc[3] = __builtin_amdgcn_wmma_f32_16x16x32_bf16(false, a_c, false, b_c3, (short)0, acc[3], false, false);
    a_c = a_n; b_c0 = b_n0; b_c1 = b_n1; b_c2 = b_n2; b_c3 = b_n3;
  }
}

// 3-wide variant for the sequential recurrence (9 tiles/wave = 3 groups of 3).
__device__ __forceinline__ void wmma_rowtile3(const unsigned short* pa,
                                              const unsigned short* pb0, const unsigned short* pb1,
                                              const unsigned short* pb2,
                                              v8f acc[3]) {
  v16bf a_c  = frag_a_at(pa, 0);
  v16bf b_c0 = frag_b_at(pb0, 0), b_c1 = frag_b_at(pb1, 0), b_c2 = frag_b_at(pb2, 0);
#pragma unroll
  for (int k0 = 0; k0 < HD; k0 += 32) {
    v16bf a_n = {}, b_n0 = {}, b_n1 = {}, b_n2 = {};
    if (k0 + 32 < HD) {
      a_n  = frag_a_at(pa, k0 + 32);
      b_n0 = frag_b_at(pb0, k0 + 32); b_n1 = frag_b_at(pb1, k0 + 32); b_n2 = frag_b_at(pb2, k0 + 32);
    }
    acc[0] = __builtin_amdgcn_wmma_f32_16x16x32_bf16(false, a_c, false, b_c0, (short)0, acc[0], false, false);
    acc[1] = __builtin_amdgcn_wmma_f32_16x16x32_bf16(false, a_c, false, b_c1, (short)0, acc[1], false, false);
    acc[2] = __builtin_amdgcn_wmma_f32_16x16x32_bf16(false, a_c, false, b_c2, (short)0, acc[2], false, false);
    a_c = a_n; b_c0 = b_n0; b_c1 = b_n1; b_c2 = b_n2;
  }
}

__global__ void k_cvt_bf16(const float* __restrict__ src, unsigned short* __restrict__ dst, int n) {
  for (int i = blockIdx.x * blockDim.x + threadIdx.x; i < n; i += gridDim.x * blockDim.x)
    dst[i] = f2bf(src[i]);
}

// Xemb[row=t*4+b][j] = emb_table[tok(t,b)][j], tok = t==0 ? input_seq[b,0] : target[b,t-1]
__global__ void k_gather_emb(const int* __restrict__ iseq, const int* __restrict__ tgt,
                             const float* __restrict__ emb, unsigned short* __restrict__ xemb) {
  for (int i = blockIdx.x * blockDim.x + threadIdx.x; i < ROWS * HD; i += gridDim.x * blockDim.x) {
    int row = i / HD, j = i - row * HD;
    int t = row >> 2, b = row & 3;
    int tok = (t == 0) ? iseq[b * TT] : tgt[b * TT + t - 1];
    xemb[i] = f2bf(emb[(size_t)tok * HD + j]);
  }
}

// h0 = tfidf[lyr] @ lyr_W^T + lyr_b   (fp32, tiny)
__global__ void k_h0(const int* __restrict__ lyr, const float* __restrict__ tfidf,
                     const float* __restrict__ lyrW, const float* __restrict__ lyrb,
                     float* __restrict__ h0) {
  int i = blockIdx.x * blockDim.x + threadIdx.x;
  if (i >= BB * HD) return;
  int b = i / HD, j = i - b * HD;
  const float* tr = tfidf + (size_t)lyr[b] * TFD;
  const float* wr = lyrW + (size_t)j * TFD;
  float s = lyrb[j];
  for (int k = 0; k < TFD; ++k) s = fmaf(tr[k], wr[k], s);
  h0[i] = s;
}

// GI = Xemb @ W_ih^T + b_ih : [2048,768] x [768,2304] -> fp32
__global__ void k_gi_gemm(const unsigned short* __restrict__ A,
                          const unsigned short* __restrict__ Bm,
                          const float* __restrict__ bias,
                          float* __restrict__ C) {
  const int lane = threadIdx.x & 31, wave = threadIdx.x >> 5;
  const int m0 = blockIdx.y * 16;
  const int job = blockIdx.x * 8 + wave;
  if (job >= (H3 / 16) / 4) return;                 // 36 wave-jobs
  const int nt0 = job * 4;
  const int nlo = lane & 15, kb16 = (lane >> 4) << 4;
  const unsigned short* pa  = A + (size_t)(m0 + nlo) * HD + ((lane >> 4) << 4);
  const unsigned short* pb0 = Bm + (size_t)(nt0 * 16 + nlo) * HD + kb16;
  const unsigned short* pb1 = pb0 + (size_t)16 * HD;
  const unsigned short* pb2 = pb0 + (size_t)32 * HD;
  const unsigned short* pb3 = pb0 + (size_t)48 * HD;
  v8f acc[4];
#pragma unroll
  for (int j = 0; j < 4; ++j) { v8f z = {}; acc[j] = z; }
  wmma_rowtile4(pa, pb0, pb1, pb2, pb3, acc);
#pragma unroll
  for (int j = 0; j < 4; ++j)
#pragma unroll
    for (int r = 0; r < 8; ++r) {
      int row = m0 + r + ((lane >> 4) << 3);
      int col = (nt0 + j) * 16 + nlo;
      C[(size_t)row * H3 + col] = acc[j][r] + bias[col];
    }
}

// Sequential GRU recurrence: one resident workgroup, 16 waves, WMMA for gh = h @ W_hh^T.
__global__ void __launch_bounds__(512) k_recurrence(
    const float* __restrict__ h0,
    const float* __restrict__ GI,            // [ROWS][H3] (b_ih folded in)
    const unsigned short* __restrict__ Whh,  // [H3][HD] bf16 (streams from L2)
    const float* __restrict__ b_hh,
    unsigned short* __restrict__ Hbf,        // [ROWS][HD] bf16 hidden states out
    float* __restrict__ hlast_out) {
  __shared__ unsigned short sh_hb[16 * HD]; // h padded to 16 rows, bf16 (A operand)
  __shared__ float sh_gh[BB * H3];
  __shared__ float sh_h[BB * HD];
  const int tid = threadIdx.x;
  const int lane = tid & 31, wave = tid >> 5;
  const int nlo = lane & 15, kb16 = (lane >> 4) << 4;

  for (int i = tid; i < 16 * HD; i += 512) {
    int m = i / HD;
    float v = (m < BB) ? h0[i] : 0.f;
    sh_hb[i] = f2bf(v);
    if (m < BB) sh_h[i] = v;
  }
  __syncthreads();

  const unsigned short* pa = sh_hb + (size_t)nlo * HD + ((lane >> 4) << 4);  // LDS lane base

  for (int t = 0; t < TT; ++t) {
    // 144 N-tiles of gh over 16 waves: 3 pipelined groups of 3 tiles each
#pragma unroll
    for (int g = 0; g < 3; ++g) {
      const int n0 = (wave * 9 + g * 3) * 16;
      const unsigned short* pb0 = Whh + (size_t)(n0 + nlo) * HD + kb16;
      const unsigned short* pb1 = pb0 + (size_t)16 * HD;
      const unsigned short* pb2 = pb0 + (size_t)32 * HD;
      v8f acc[3];
      { v8f z = {}; acc[0] = z; acc[1] = z; acc[2] = z; }
      wmma_rowtile3(pa, pb0, pb1, pb2, acc);
#pragma unroll
      for (int j = 0; j < 3; ++j)
#pragma unroll
        for (int r = 0; r < 8; ++r) {
          int m = r + ((lane >> 4) << 3);
          if (m < BB) sh_gh[m * H3 + n0 + j * 16 + nlo] = acc[j][r];
        }
    }
    __syncthreads();

    const float* gi = GI + (size_t)(t * BB) * H3;
    for (int i = tid; i < BB * HD; i += 512) {
      int b = i / HD, j = i - b * HD;
      float ir = gi[b * H3 + j];
      float iz = gi[b * H3 + HD + j];
      float in = gi[b * H3 + 2 * HD + j];
      float hr = sh_gh[b * H3 + j] + b_hh[j];
      float hz = sh_gh[b * H3 + HD + j] + b_hh[HD + j];
      float hn = sh_gh[b * H3 + 2 * HD + j] + b_hh[2 * HD + j];
      float r = 1.f / (1.f + __expf(-(ir + hr)));
      float z = 1.f / (1.f + __expf(-(iz + hz)));
      float n = tanhf(in + r * hn);
      float hnew = (1.f - z) * n + z * sh_h[i];    // ctx == h (softmax over singleton)
      sh_h[i] = hnew;
      unsigned short hb = f2bf(hnew);
      sh_hb[i] = hb;
      Hbf[(size_t)(t * BB + b) * HD + j] = hb;
    }
    __syncthreads();
  }
  for (int i = tid; i < BB * HD; i += 512) hlast_out[i] = sh_h[i];  // [1,B,H]
}

// logits = Hmat @ out_W^T + out_b, written directly as dec_out[b][t][v]
__global__ void k_out_gemm(const unsigned short* __restrict__ A,
                           const unsigned short* __restrict__ Bm,
                           const float* __restrict__ bias,
                           float* __restrict__ out) {
  const int lane = threadIdx.x & 31, wave = threadIdx.x >> 5;
  const int m0 = blockIdx.y * 16;
  const int job = blockIdx.x * 8 + wave;
  if (job >= (VOC / 16) / 4) return;                // 500 wave-jobs
  const int nt0 = job * 4;
  __builtin_prefetch(Bm + (size_t)nt0 * 16 * HD, 0, 1);   // global_prefetch_b8
  const int nlo = lane & 15, kb16 = (lane >> 4) << 4;
  const unsigned short* pa  = A + (size_t)(m0 + nlo) * HD + ((lane >> 4) << 4);
  const unsigned short* pb0 = Bm + (size_t)(nt0 * 16 + nlo) * HD + kb16;
  const unsigned short* pb1 = pb0 + (size_t)16 * HD;
  const unsigned short* pb2 = pb0 + (size_t)32 * HD;
  const unsigned short* pb3 = pb0 + (size_t)48 * HD;
  v8f acc[4];
#pragma unroll
  for (int j = 0; j < 4; ++j) { v8f z = {}; acc[j] = z; }
  wmma_rowtile4(pa, pb0, pb1, pb2, pb3, acc);
#pragma unroll
  for (int j = 0; j < 4; ++j)
#pragma unroll
    for (int r = 0; r < 8; ++r) {
      int row = m0 + r + ((lane >> 4) << 3);
      int t = row >> 2, b = row & 3;
      int col = (nt0 + j) * 16 + nlo;
      out[((size_t)b * TT + t) * VOC + col] = acc[j][r] + bias[col];
    }
}

// In-place log_softmax over V per (b,t) row. (Second log_softmax in ref is the identity.)
__global__ void k_logsoftmax(float* __restrict__ out) {
  float* p = out + (size_t)blockIdx.x * VOC;
  __shared__ float red[256];
  const int tid = threadIdx.x;
  float mx = -3.4e38f;
  for (int i = tid; i < VOC; i += 256) mx = fmaxf(mx, p[i]);
  red[tid] = mx; __syncthreads();
  for (int s = 128; s > 0; s >>= 1) { if (tid < s) red[tid] = fmaxf(red[tid], red[tid + s]); __syncthreads(); }
  mx = red[0]; __syncthreads();
  float sum = 0.f;
  for (int i = tid; i < VOC; i += 256) sum += __expf(p[i] - mx);
  red[tid] = sum; __syncthreads();
  for (int s = 128; s > 0; s >>= 1) { if (tid < s) red[tid] += red[tid + s]; __syncthreads(); }
  float lse = mx + __logf(red[0]);
  for (int i = tid; i < VOC; i += 256) p[i] -= lse;
}

// attentions == softmax over a singleton == 1.0 everywhere
__global__ void k_attn(float* __restrict__ out) {
  int i = blockIdx.x * blockDim.x + threadIdx.x;
  if (i < BB * TT) out[i] = 1.0f;
}

extern "C" void kernel_launch(void* const* d_in, const int* in_sizes, int n_in,
                              void* d_out, int out_size, void* d_ws, size_t ws_size,
                              hipStream_t stream) {
  (void)in_sizes; (void)n_in; (void)out_size; (void)ws_size;
  const int*   lyr   = (const int*)d_in[0];
  const int*   iseq  = (const int*)d_in[1];
  const int*   tgt   = (const int*)d_in[2];
  const float* emb   = (const float*)d_in[3];
  // d_in[4..9]: Wa/ba/Ua/ub/Va/vb are dead code (softmax over singleton => w==1, ctx==h)
  const float* tfidf = (const float*)d_in[10];
  const float* lyrW  = (const float*)d_in[11];
  const float* lyrb  = (const float*)d_in[12];
  const float* W_ih  = (const float*)d_in[13];
  const float* W_hh  = (const float*)d_in[14];
  const float* b_ih  = (const float*)d_in[15];
  const float* b_hh  = (const float*)d_in[16];
  const float* out_W = (const float*)d_in[17];
  const float* out_b = (const float*)d_in[18];
  float* out = (float*)d_out;

  char* ws = (char*)d_ws;
  size_t off = 0;
  auto take = [&](size_t bytes) { char* p = ws + off; off = (off + bytes + 255) & ~(size_t)255; return p; };
  unsigned short* wih_bf  = (unsigned short*)take(sizeof(unsigned short) * (size_t)H3 * HD);
  unsigned short* whh_bf  = (unsigned short*)take(sizeof(unsigned short) * (size_t)H3 * HD);
  unsigned short* outw_bf = (unsigned short*)take(sizeof(unsigned short) * (size_t)VOC * HD);
  unsigned short* xemb_bf = (unsigned short*)take(sizeof(unsigned short) * (size_t)ROWS * HD);
  unsigned short* hmat_bf = (unsigned short*)take(sizeof(unsigned short) * (size_t)ROWS * HD);
  float* GI = (float*)take(sizeof(float) * (size_t)ROWS * H3);
  float* H0 = (float*)take(sizeof(float) * (size_t)BB * HD);

  k_cvt_bf16<<<dim3(2048), dim3(256), 0, stream>>>(W_ih, wih_bf, H3 * HD);
  k_cvt_bf16<<<dim3(2048), dim3(256), 0, stream>>>(W_hh, whh_bf, H3 * HD);
  k_cvt_bf16<<<dim3(8192), dim3(256), 0, stream>>>(out_W, outw_bf, VOC * HD);
  k_gather_emb<<<dim3(2048), dim3(256), 0, stream>>>(iseq, tgt, emb, xemb_bf);
  k_h0<<<dim3(12), dim3(256), 0, stream>>>(lyr, tfidf, lyrW, lyrb, H0);
  k_gi_gemm<<<dim3(5, ROWS / 16), dim3(256), 0, stream>>>(xemb_bf, wih_bf, b_ih, GI);

  const size_t DEC = (size_t)BB * TT * VOC;   // 65,536,000
  k_recurrence<<<dim3(1), dim3(512), 0, stream>>>(H0, GI, whh_bf, b_hh, hmat_bf, out + DEC);
  k_out_gemm<<<dim3(63, ROWS / 16), dim3(256), 0, stream>>>(hmat_bf, outw_bf, out_b, out);
  k_logsoftmax<<<dim3(ROWS), dim3(256), 0, stream>>>(out);
  k_attn<<<dim3(8), dim3(256), 0, stream>>>(out + DEC + (size_t)BB * HD);
}
